// GraphAttentionLayer_topk_867583394402
// MI455X (gfx1250) — matrix operand order repaired and checked by reading, compiled, not run on hardware
//
#include <hip/hip_runtime.h>
#include <hip/hip_bf16.h>
#include <math.h>
#include <stdint.h>

typedef __attribute__((ext_vector_type(16))) _Float16 v16h;
typedef __attribute__((ext_vector_type(8)))  _Float16 v8h;
typedef __attribute__((ext_vector_type(4)))  _Float16 v4h;
typedef __attribute__((ext_vector_type(8)))  float    v8f;

#define F_DIM   512
#define N_NODES 2048
#define M_TOTAL 8192          // B * N
#define NEG_BIG (-9.0e15f)
#define ALPHA_LRELU 0.2f
#define KSEL 16

__device__ __forceinline__ void async_copy_16B_to_lds(unsigned ldsOff, const void* gsrc) {
  // GLOBAL_LOAD_ASYNC_TO_LDS_B128 (GV mode): copies 16 B global -> LDS at ldsOff, counted by ASYNCcnt
  asm volatile("global_load_async_to_lds_b128 %0, %1, off"
               :: "v"(ldsOff), "v"(gsrc) : "memory");
}

__device__ __forceinline__ void wait_async_zero() {
#if __has_builtin(__builtin_amdgcn_s_wait_asynccnt)
  __builtin_amdgcn_s_wait_asynccnt(0);
#else
  asm volatile("s_wait_asynccnt 0" ::: "memory");
#endif
}

// ---------------------------------------------------------------------------
// Kernel 0a: h (f32) -> h16 (f16), vectorized 4-wide.
// ---------------------------------------------------------------------------
__global__ __launch_bounds__(256)
void gat_cvt_h(const float* __restrict__ h, _Float16* __restrict__ h16) {
  const size_t i = ((size_t)blockIdx.x * 256 + threadIdx.x) * 4;
  float4 v = *(const float4*)(h + i);
  v4h o = { (_Float16)v.x, (_Float16)v.y, (_Float16)v.z, (_Float16)v.w };
  *(v4h*)(h16 + i) = o;
}

// ---------------------------------------------------------------------------
// Kernel 0b: W (f32, k-major) -> Wt16 (f16, n-major transpose), LDS-tiled.
// ---------------------------------------------------------------------------
__global__ __launch_bounds__(256)
void gat_cvt_wt(const float* __restrict__ W, _Float16* __restrict__ Wt) {
  __shared__ _Float16 tile[64][65];
  const int tid = threadIdx.x;
  const int k0 = blockIdx.x * 64, n0 = blockIdx.y * 64;
  for (int i = tid; i < 64 * 64; i += 256) {
    int r = i >> 6, c = i & 63;
    tile[r][c] = (_Float16)W[(size_t)(k0 + r) * F_DIM + n0 + c];
  }
  __syncthreads();
  for (int i = tid; i < 64 * 64; i += 256) {
    int r = i >> 6, c = i & 63;     // write coalesced along k
    Wt[(size_t)(n0 + r) * F_DIM + k0 + c] = tile[c][r];
  }
}

// ---------------------------------------------------------------------------
// Kernel 1: Wh = h @ W  (8192x512 @ 512x512), f16 WMMA, f32 accumulate.
// Block = 256 threads (8 waves). Block tile 128(M) x 64(N), K step 32.
// Staging uses GLOBAL_LOAD_ASYNC_TO_LDS_B128 (no VGPR round-trip), fragments
// are contiguous b128 LDS runs per ISA 7.12.2.
// ---------------------------------------------------------------------------
__global__ __launch_bounds__(256)
void gat_gemm_wh(const _Float16* __restrict__ h16, const _Float16* __restrict__ Wt16,
                 float* __restrict__ Wh) {
  __shared__ _Float16 As[128][32];   // M-major, 8 KB
  __shared__ _Float16 Bs[64][32];    // N-major (from transposed W), 4 KB

  const int tid  = threadIdx.x;
  const int wave = tid >> 5;
  const int lane = tid & 31;
  const int rowBase = blockIdx.x * 128;
  const int colBase = blockIdx.y * 64;

  v8f acc0 = {}, acc1 = {}, acc2 = {}, acc3 = {};

  const int mLoc = lane & 15;
  const int hi   = lane >> 4;
  const int kbA  = hi * 8;     // A: lanes16-31 hold the K+8 run
  const int kbB  = hi * 8;     // B: lanes16-31 hold the K+8 run

  // Per-thread staging coordinates (fixed across K steps)
  const int ra0 = tid >> 2,          ca = (tid & 3) * 8;      // A pass 0
  const int ra1 = (tid + 256) >> 2;                           // A pass 1
  const int rb  = tid >> 2;                                   // B (cb == ca)
  const unsigned ldsA0 = (unsigned)(uintptr_t)&As[ra0][ca];
  const unsigned ldsA1 = (unsigned)(uintptr_t)&As[ra1][ca];
  const unsigned ldsB  = (unsigned)(uintptr_t)&Bs[rb][ca];

  for (int k0 = 0; k0 < F_DIM; k0 += 32) {
    async_copy_16B_to_lds(ldsA0, &h16[(size_t)(rowBase + ra0) * F_DIM + k0 + ca]);
    async_copy_16B_to_lds(ldsA1, &h16[(size_t)(rowBase + ra1) * F_DIM + k0 + ca]);
    async_copy_16B_to_lds(ldsB,  &Wt16[(size_t)(colBase + rb) * F_DIM + k0 + ca]);
    wait_async_zero();
    __syncthreads();

    const int ar = wave * 16 + mLoc;
    v8h alo = *(const v8h*)&As[ar][kbA];
    v8h ahi = *(const v8h*)&As[ar][16 + kbA];
    v16h afrag = __builtin_shufflevector(alo, ahi,
        0,1,2,3,4,5,6,7,8,9,10,11,12,13,14,15);

#pragma unroll
    for (int t = 0; t < 4; ++t) {
      const int br = t * 16 + mLoc;
      v8h blo = *(const v8h*)&Bs[br][kbB];
      v8h bhi = *(const v8h*)&Bs[br][16 + kbB];
      v16h bfrag = __builtin_shufflevector(blo, bhi,
          0,1,2,3,4,5,6,7,8,9,10,11,12,13,14,15);
      if      (t == 0) acc0 = __builtin_amdgcn_wmma_f32_16x16x32_f16(false, afrag, false, bfrag, (short)0, acc0, false, false);
      else if (t == 1) acc1 = __builtin_amdgcn_wmma_f32_16x16x32_f16(false, afrag, false, bfrag, (short)0, acc1, false, false);
      else if (t == 2) acc2 = __builtin_amdgcn_wmma_f32_16x16x32_f16(false, afrag, false, bfrag, (short)0, acc2, false, false);
      else             acc3 = __builtin_amdgcn_wmma_f32_16x16x32_f16(false, afrag, false, bfrag, (short)0, acc3, false, false);
    }
    __syncthreads();
  }

  // Store C: VGPR r -> row (hi*8 + r), col = mLoc (ISA 32-bit C/D layout)
  const int outRow = rowBase + wave * 16 + hi * 8;
#pragma unroll
  for (int r = 0; r < 8; ++r) {
    float* o = Wh + (size_t)(outRow + r) * F_DIM + colBase + mLoc;
    o[0]  = acc0[r];
    o[16] = acc1[r];
    o[32] = acc2[r];
    o[48] = acc3[r];
  }
}

// ---------------------------------------------------------------------------
// Kernel 2: e_src[i] = Wh[i,:]·a1 ; e_dst[i] = Wh[i,:]·a2. One wave per row.
// ---------------------------------------------------------------------------
__global__ __launch_bounds__(256)
void gat_edge_logits(const float* __restrict__ Wh, const float* __restrict__ a,
                     float* __restrict__ e_src, float* __restrict__ e_dst) {
  const int row  = blockIdx.x * 8 + (threadIdx.x >> 5);
  const int lane = threadIdx.x & 31;
  const float* wr = Wh + (size_t)row * F_DIM;
  float s1 = 0.f, s2 = 0.f;
  for (int f = lane; f < F_DIM; f += 32) {
    float v = wr[f];
    s1 += v * a[f];
    s2 += v * a[F_DIM + f];
  }
#pragma unroll
  for (int off = 16; off > 0; off >>= 1) {
    s1 += __shfl_xor(s1, off, 32);
    s2 += __shfl_xor(s2, off, 32);
  }
  if (lane == 0) { e_src[row] = s1; e_dst[row] = s2; }
}

// ---------------------------------------------------------------------------
// Kernel 3 (fused): per row: e-row (leakyrelu + adj mask), softmax stats,
// 16 argmax rounds -> kth threshold, compact survivors, sparse gather + ELU.
// One 256-thread block per row; adj read exactly once (HBM roofline pass).
// ---------------------------------------------------------------------------
__global__ __launch_bounds__(256)
void gat_attn_topk(const float* __restrict__ Wh, const int* __restrict__ adj,
                   const float* __restrict__ e_src, const float* __restrict__ e_dst,
                   float* __restrict__ out) {
  const int row = blockIdx.x;          // 0..8191  (= b*2048 + n)
  const int b   = row >> 11;
  const int tid = threadIdx.x;

  __shared__ float eRow[N_NODES];
  __shared__ float eWork[N_NODES];
  __shared__ float red[256];
  __shared__ int   redIdx[256];
  __shared__ float sMax, sSum, sKth;
  __shared__ int   sCount;
  __shared__ int   sIdx[64];
  __shared__ float sW[64];

  const float es = e_src[row];
  const int* adjRow = adj + (size_t)row * N_NODES;
  const float* eD = e_dst + ((size_t)b << 11);

  float lmax = -INFINITY;
  for (int j = tid; j < N_NODES; j += 256) {
    float e = es + eD[j];
    e = e > 0.f ? e : ALPHA_LRELU * e;
    if (adjRow[j] <= 0) e = NEG_BIG;
    eRow[j]  = e;
    eWork[j] = e;
    lmax = fmaxf(lmax, e);
  }
  red[tid] = lmax;
  __syncthreads();
  for (int s = 128; s > 0; s >>= 1) {
    if (tid < s) red[tid] = fmaxf(red[tid], red[tid + s]);
    __syncthreads();
  }
  if (tid == 0) sMax = red[0];
  __syncthreads();
  const float rmax = sMax;

  float lsum = 0.f;
  for (int j = tid; j < N_NODES; j += 256) lsum += __expf(eRow[j] - rmax);
  red[tid] = lsum;
  __syncthreads();
  for (int s = 128; s > 0; s >>= 1) {
    if (tid < s) red[tid] += red[tid + s];
    __syncthreads();
  }
  if (tid == 0) sSum = red[0];
  __syncthreads();
  const float inv = 1.0f / sSum;

  for (int it = 0; it < KSEL; ++it) {
    float lm = -INFINITY; int li = -1;
    for (int j = tid; j < N_NODES; j += 256) {
      float v = eWork[j];
      if (v > lm) { lm = v; li = j; }
    }
    red[tid] = lm; redIdx[tid] = li;
    __syncthreads();
    for (int s = 128; s > 0; s >>= 1) {
      if (tid < s && red[tid + s] > red[tid]) {
        red[tid] = red[tid + s]; redIdx[tid] = redIdx[tid + s];
      }
      __syncthreads();
    }
    if (tid == 0) {
      sKth = red[0];
      if (redIdx[0] >= 0) eWork[redIdx[0]] = -INFINITY;
      sCount = 0;
    }
    __syncthreads();
  }
  const float kth = sKth;   // survivors: e >= kth (ties kept, matches ref)

  for (int j = tid; j < N_NODES; j += 256) {
    if (eRow[j] >= kth) {
      int p = atomicAdd(&sCount, 1);
      if (p < 64) { sIdx[p] = j; sW[p] = __expf(eRow[j] - rmax) * inv; }
    }
  }
  __syncthreads();
  const int cnt = min(sCount, 64);

  float acc0 = 0.f, acc1 = 0.f;
  const int f0 = tid, f1 = tid + 256;
  for (int s = 0; s < cnt; ++s) {
    const float* wr = Wh + (((size_t)b << 11) + sIdx[s]) * F_DIM;
    const float w = sW[s];
    acc0 += w * wr[f0];
    acc1 += w * wr[f1];
  }
  float* o = out + (size_t)row * F_DIM;
  o[f0] = acc0 > 0.f ? acc0 : __expf(acc0) - 1.0f;
  o[f1] = acc1 > 0.f ? acc1 : __expf(acc1) - 1.0f;
}

// ---------------------------------------------------------------------------
extern "C" void kernel_launch(void* const* d_in, const int* in_sizes, int n_in,
                              void* d_out, int out_size, void* d_ws, size_t ws_size,
                              hipStream_t stream) {
  const float* h   = (const float*)d_in[0];   // [4,2048,512]
  const int*   adj = (const int*)  d_in[1];   // [4,2048,2048]
  const float* W   = (const float*)d_in[2];   // [512,512]
  const float* a   = (const float*)d_in[3];   // [1024,1]
  float* out = (float*)d_out;                 // [4,2048,512]

  float*    Wh    = (float*)d_ws;                      // 4M f32 = 16 MB
  float*    e_src = Wh + (size_t)M_TOTAL * F_DIM;      // 8192 f32
  float*    e_dst = e_src + M_TOTAL;                   // 8192 f32
  _Float16* h16   = (_Float16*)(e_dst + M_TOTAL);      // 4M f16 = 8 MB (16B aligned)
  _Float16* Wt16  = h16 + (size_t)M_TOTAL * F_DIM;     // 256K f16 = 512 KB

  gat_cvt_h<<<(M_TOTAL * F_DIM) / (256 * 4), 256, 0, stream>>>(h, h16);
  gat_cvt_wt<<<dim3(F_DIM / 64, F_DIM / 64), 256, 0, stream>>>(W, Wt16);

  dim3 g1(M_TOTAL / 128, F_DIM / 64);   // (64, 8)
  gat_gemm_wh<<<g1, 256, 0, stream>>>(h16, Wt16, Wh);

  gat_edge_logits<<<M_TOTAL / 8, 256, 0, stream>>>(Wh, a, e_src, e_dst);

  gat_attn_topk<<<M_TOTAL, 256, 0, stream>>>(Wh, adj, e_src, e_dst, out);
}
